// MutilCrossAttention_44856638439715
// MI455X (gfx1250) — matrix operand order repaired and checked by reading, not compiled, not run on hardware
//
#include <hip/hip_runtime.h>
#include <hip/hip_bf16.h>

// ---------------- types ----------------
typedef __attribute__((ext_vector_type(16))) _Float16 v16h;
typedef __attribute__((ext_vector_type(8)))  _Float16 v8h;
typedef __attribute__((ext_vector_type(8)))  float    v8f;

#define NB   512          // batch
#define NH   8            // heads
#define DD   1024         // per-head dim
#define SQRT_D 32.0f

// ---------------- fragment helpers (contiguous k order, same for A and B) ----------------
// A-matrix 16x32 f16: lane<16 -> K[0..15]; lane>=16 -> K[16..31], contiguous per lane
__device__ __forceinline__ v16h load_a_frag(const _Float16* p, int hi) {
    v8h lo = *(const v8h*)(p + hi * 16);
    v8h hb = *(const v8h*)(p + hi * 16 + 8);
    v16h r;
#pragma unroll
    for (int i = 0; i < 8; ++i) { r[i] = lo[i]; r[i + 8] = hb[i]; }
    return r;
}
// B-matrix 32x16 f16: lane<16 -> K[0..15]; lane>=16 -> K[16..31], contiguous per lane
__device__ __forceinline__ v16h load_b_frag(const _Float16* p, int hi) {
    const _Float16* q = p + hi * 16;
    v8h lo = *(const v8h*)(q);
    v8h hb = *(const v8h*)(q + 8);
    v16h r;
#pragma unroll
    for (int i = 0; i < 8; ++i) { r[i] = lo[i]; r[i + 8] = hb[i]; }
    return r;
}
__device__ __forceinline__ v8f vzero8() {
    v8f z;
#pragma unroll
    for (int i = 0; i < 8; ++i) z[i] = 0.0f;
    return z;
}
__device__ __forceinline__ v8f wmma16(v16h a, v16h b, v8f c) {
    return __builtin_amdgcn_wmma_f32_16x16x32_f16(false, a, false, b, (short)0, c,
                                                  false, false);
}

// ---------------- conversion kernels ----------------
// in: fp32 [n][h][d] (n*8192 + h*1024 + d)  ->  out: f16 [h][n][d]
__global__ __launch_bounds__(256) void cvt_qkv_kernel(const float* __restrict__ in,
                                                      _Float16* __restrict__ out) {
    int idx = blockIdx.x * 256 + threadIdx.x;
    if (idx >= NB * NH * DD) return;
    int d = idx & (DD - 1);
    int n = (idx >> 10) & (NB - 1);
    int h = idx >> 19;
    out[idx] = (_Float16)in[(size_t)n * (NH * DD) + (size_t)h * DD + d];
}

// straight fp32 -> f16 copy (weights keep [h][e][d] layout)
__global__ __launch_bounds__(256) void cvt_w_kernel(const float* __restrict__ in,
                                                    _Float16* __restrict__ out, int n) {
    int idx = blockIdx.x * 256 + threadIdx.x;
    if (idx < n) out[idx] = (_Float16)in[idx];
}

// ---------------- projection: Y[h] = X[h] @ W[h]^T + b[h] ----------------
// X: f16 [h][512][1024]; W: f16 [h][e=1024][d=1024]; bias fp32 [h][1024]
// transposed==0 -> Y f16 [h][512][1024];  transposed!=0 -> Y f16 [h][1024][512]
// Each wave computes a 32x64 output tile (2 row-tiles x 4 col-tiles), software
// pipelined with a 2x-unrolled ping-pong so buffer rotation is register-renamed
// (no v_mov chatter) and each load group is covered by a full WMMA group.
__global__ __launch_bounds__(256) void proj_kernel(const _Float16* __restrict__ X,
                                                   const _Float16* __restrict__ W,
                                                   const float* __restrict__ bias,
                                                   _Float16* __restrict__ Y,
                                                   int transposed) {
    const int tid  = threadIdx.x;
    const int lane = tid & 31;
    const int wave = tid >> 5;
    const int r    = lane & 15;
    const int hi   = lane >> 4;

    int job = blockIdx.x * 8 + wave;      // 8 heads * 16 mgroups * 16 egroups = 2048
    const int eg   = job & 15; job >>= 4;
    const int mg   = job & 15; job >>= 4;
    const int head = job;

    const _Float16* Xh = X + (size_t)head * NB * DD;
    const _Float16* Wh = W + (size_t)head * DD * DD;
    const float*    bh = bias + head * DD;

    const int m0 = mg * 32;
    const int e0 = eg * 64;

    v8f acc[2][4];
#pragma unroll
    for (int s = 0; s < 2; ++s)
#pragma unroll
        for (int t = 0; t < 4; ++t) acc[s][t] = vzero8();

    const _Float16* a0p = Xh + (size_t)(m0 + r) * DD;
    const _Float16* a1p = a0p + (size_t)16 * DD;
    const _Float16* bp[4];
#pragma unroll
    for (int t = 0; t < 4; ++t) bp[t] = Wh + (size_t)(e0 + t * 16 + r) * DD;

    // prologue: K-step 0 fragments
    v16h a0 = load_a_frag(a0p, hi);
    v16h a1 = load_a_frag(a1p, hi);
    v16h b[4];
#pragma unroll
    for (int t = 0; t < 4; ++t) b[t] = load_b_frag(bp[t], hi);

#pragma unroll 2
    for (int k0 = 0; k0 < DD; k0 += 32) {
        const int kn = (k0 + 32) & (DD - 1);   // wraps on last iter (harmless)
        // issue next-step loads first (distinct registers -> overlap with WMMA)
        v16h a0n = load_a_frag(a0p + kn, hi);
        v16h a1n = load_a_frag(a1p + kn, hi);
        v16h bn[4];
#pragma unroll
        for (int t = 0; t < 4; ++t) bn[t] = load_b_frag(bp[t] + kn, hi);
        // deep prefetch of the weight streams (global_prefetch_b8)
        const int kp = (k0 + 256) & (DD - 1);
#pragma unroll
        for (int t = 0; t < 4; ++t) __builtin_prefetch(bp[t] + kp, 0, 3);

#pragma unroll
        for (int t = 0; t < 4; ++t) {
            acc[0][t] = wmma16(a0, b[t], acc[0][t]);
            acc[1][t] = wmma16(a1, b[t], acc[1][t]);
        }
        a0 = a0n; a1 = a1n;
#pragma unroll
        for (int t = 0; t < 4; ++t) b[t] = bn[t];
    }

#pragma unroll
    for (int t = 0; t < 4; ++t) {
        const int e  = e0 + t * 16 + r;      // lane's output column
        const float bv = bh[e];
#pragma unroll
        for (int s = 0; s < 2; ++s) {
#pragma unroll
            for (int g = 0; g < 8; ++g) {
                const int m = m0 + s * 16 + g + hi * 8;  // C/D frag row
                const float val = acc[s][t][g] + bv;
                if (!transposed)
                    Y[(size_t)head * NB * DD + (size_t)m * DD + e] = (_Float16)val;
                else
                    Y[(size_t)head * DD * NB + (size_t)e * NB + m] = (_Float16)val;
            }
        }
    }
}

// ---------------- attention: per (head, 16-row block) ----------------
// Qp,Kp: f16 [h][512][1024]; Vpt: f16 [h][1024][512]; Out: fp32 [512][8][1024]
__global__ __launch_bounds__(256) void attn_kernel(const _Float16* __restrict__ Qp,
                                                   const _Float16* __restrict__ Kp,
                                                   const _Float16* __restrict__ Vpt,
                                                   float* __restrict__ Out) {
    __shared__ float    S[16][NB];     // 32 KB f32 scores
    __shared__ _Float16 P[16][NB];     // 16 KB f16 probabilities
    __shared__ float    rmax[16], rinv[16];

    const int tid  = threadIdx.x;
    const int lane = tid & 31;
    const int wave = tid >> 5;
    const int r    = lane & 15;
    const int hi   = lane >> 4;

    const int head = blockIdx.x >> 5;  // 32 n-tiles per head
    const int n0   = (blockIdx.x & 31) * 16;

    const _Float16* Qh = Qp  + (size_t)head * NB * DD;
    const _Float16* Kh = Kp  + (size_t)head * NB * DD;
    const _Float16* Vh = Vpt + (size_t)head * DD * NB;

    // ---- phase 1: S = (Q Kt) * sqrt(d); each wave covers 4 of 32 m-tiles,
    //      processed in pairs so each A-fragment feeds 2 WMMAs; pipelined ----
    const _Float16* qrow = Qh + (size_t)(n0 + r) * DD;
#pragma unroll
    for (int pair = 0; pair < 2; ++pair) {
        const int m0a = (wave + pair * 16) * 16;
        const int m0b = m0a + 8 * 16;
        const _Float16* kra = Kh + (size_t)(m0a + r) * DD;
        const _Float16* krb = Kh + (size_t)(m0b + r) * DD;
        v8f acc0 = vzero8(), acc1 = vzero8();
        v16h a  = load_a_frag(qrow, hi);
        v16h b0 = load_b_frag(kra, hi);
        v16h b1 = load_b_frag(krb, hi);
#pragma unroll 2
        for (int k0 = 0; k0 < DD; k0 += 32) {
            const int kn = (k0 + 32) & (DD - 1);
            v16h an  = load_a_frag(qrow + kn, hi);
            v16h bn0 = load_b_frag(kra + kn, hi);
            v16h bn1 = load_b_frag(krb + kn, hi);
            acc0 = wmma16(a, b0, acc0);
            acc1 = wmma16(a, b1, acc1);
            a = an; b0 = bn0; b1 = bn1;
        }
#pragma unroll
        for (int g = 0; g < 8; ++g) {
            S[g + hi * 8][m0a + r] = acc0[g] * SQRT_D;
            S[g + hi * 8][m0b + r] = acc1[g] * SQRT_D;
        }
    }
    __syncthreads();

    // ---- phase 2: softmax over 512 cols; wave w owns rows 2w, 2w+1 ----
    for (int rr = wave * 2; rr < wave * 2 + 2; ++rr) {
        float mx = -1e30f;
        for (int c = lane; c < NB; c += 32) mx = fmaxf(mx, S[rr][c]);
#pragma unroll
        for (int off = 16; off; off >>= 1) mx = fmaxf(mx, __shfl_xor(mx, off, 32));
        float sm = 0.0f;
        for (int c = lane; c < NB; c += 32) sm += __expf(S[rr][c] - mx);
#pragma unroll
        for (int off = 16; off; off >>= 1) sm += __shfl_xor(sm, off, 32);
        if (lane == 0) { rmax[rr] = mx; rinv[rr] = 1.0f / sm; }
    }
    __syncthreads();
    for (int idx = tid; idx < 16 * NB; idx += 256) {
        const int rr = idx >> 9, c = idx & (NB - 1);
        P[rr][c] = (_Float16)(__expf(S[rr][c] - rmax[rr]) * rinv[rr]);
    }
    __syncthreads();

    // ---- phase 3: O = P @ V; each wave owns 8 of 64 e-tiles, in pairs ----
    const _Float16* prow = &P[r][0];   // A-frags straight from LDS (ds_load_b128)
#pragma unroll
    for (int pair = 0; pair < 4; ++pair) {
        const int e0a = (wave * 8 + pair * 2) * 16;
        const int e0b = e0a + 16;
        const _Float16* vra = Vh + (size_t)(e0a + r) * NB;
        const _Float16* vrb = Vh + (size_t)(e0b + r) * NB;
        v8f acc0 = vzero8(), acc1 = vzero8();
        v16h a  = load_a_frag(prow, hi);
        v16h b0 = load_b_frag(vra, hi);
        v16h b1 = load_b_frag(vrb, hi);
#pragma unroll 2
        for (int k0 = 0; k0 < NB; k0 += 32) {
            const int kn = (k0 + 32) & (NB - 1);
            v16h an  = load_a_frag(prow + kn, hi);
            v16h bn0 = load_b_frag(vra + kn, hi);
            v16h bn1 = load_b_frag(vrb + kn, hi);
            acc0 = wmma16(a, b0, acc0);
            acc1 = wmma16(a, b1, acc1);
            a = an; b0 = bn0; b1 = bn1;
        }
#pragma unroll
        for (int g = 0; g < 8; ++g) {
            const int n = n0 + g + hi * 8;
            Out[(size_t)n * (NH * DD) + (size_t)head * DD + e0a + r] = acc0[g];
            Out[(size_t)n * (NH * DD) + (size_t)head * DD + e0b + r] = acc1[g];
        }
    }
}

// ---------------- host launcher ----------------
extern "C" void kernel_launch(void* const* d_in, const int* in_sizes, int n_in,
                              void* d_out, int out_size, void* d_ws, size_t ws_size,
                              hipStream_t stream) {
    (void)in_sizes; (void)n_in; (void)out_size; (void)ws_size;

    const float* q  = (const float*)d_in[0];
    const float* k  = (const float*)d_in[1];
    const float* v  = (const float*)d_in[2];
    const float* W1 = (const float*)d_in[3];
    const float* b1 = (const float*)d_in[4];
    const float* W2 = (const float*)d_in[5];
    const float* b2 = (const float*)d_in[6];
    const float* W3 = (const float*)d_in[7];
    const float* b3 = (const float*)d_in[8];
    float* out = (float*)d_out;

    const size_t S1 = (size_t)NH * NB * DD * sizeof(_Float16);   //  8 MiB
    const size_t SW = (size_t)NH * DD * DD * sizeof(_Float16);   // 16 MiB
    char* ws = (char*)d_ws;
    _Float16* q16 = (_Float16*)(ws);
    _Float16* k16 = (_Float16*)(ws + S1);
    _Float16* v16 = (_Float16*)(ws + 2 * S1);
    _Float16* w1h = (_Float16*)(ws + 3 * S1);
    _Float16* w2h = (_Float16*)(ws + 3 * S1 + SW);
    _Float16* w3h = (_Float16*)(ws + 3 * S1 + 2 * SW);
    _Float16* Qp  = (_Float16*)(ws + 3 * S1 + 3 * SW);
    _Float16* Kp  = (_Float16*)(ws + 4 * S1 + 3 * SW);
    _Float16* Vpt = (_Float16*)(ws + 5 * S1 + 3 * SW);

    const int nAct = NB * NH * DD;           // 4,194,304
    const int nW   = NH * DD * DD;           // 8,388,608
    dim3 blk(256);

    cvt_qkv_kernel<<<(nAct + 255) / 256, blk, 0, stream>>>(q, q16);
    cvt_qkv_kernel<<<(nAct + 255) / 256, blk, 0, stream>>>(k, k16);
    cvt_qkv_kernel<<<(nAct + 255) / 256, blk, 0, stream>>>(v, v16);
    cvt_w_kernel<<<(nW + 255) / 256, blk, 0, stream>>>(W1, w1h, nW);
    cvt_w_kernel<<<(nW + 255) / 256, blk, 0, stream>>>(W2, w2h, nW);
    cvt_w_kernel<<<(nW + 255) / 256, blk, 0, stream>>>(W3, w3h, nW);

    // 8 heads * 16 m-groups(32 rows) * 16 e-groups(64 cols) = 2048 wave-jobs
    proj_kernel<<<256, blk, 0, stream>>>(q16, w1h, b1, Qp, 0);
    proj_kernel<<<256, blk, 0, stream>>>(k16, w2h, b2, Kp, 0);
    proj_kernel<<<256, blk, 0, stream>>>(v16, w3h, b3, Vpt, 1);  // store [h][e][n]

    attn_kernel<<<NH * (NB / 16), blk, 0, stream>>>(Qp, Kp, Vpt, out);
}
